// TbNetV1_8864812499411
// MI455X (gfx1250) — matrix operand order repaired and checked by reading, not compiled, not run on hardware
//
#include <hip/hip_runtime.h>
#include <hip/hip_bf16.h>
#include <math.h>

#define N_NODES 20000
#define N_EDGES 320000
#define LSEQ 16

typedef __attribute__((ext_vector_type(16))) _Float16 v16h;
typedef __attribute__((ext_vector_type(8)))  _Float16 v8h;
typedef __attribute__((ext_vector_type(8)))  float    v8f;

__device__ __forceinline__ v8f wmma16x16x32(v16h a, v16h b, v8f c) {
  // D = A(16x32 f16) * B(32x16 f16) + C(f32)
  return __builtin_amdgcn_wmma_f32_16x16x32_f16(false, a, false, b, (short)0, c,
                                                false, false);
}

// A fragment (16x32 f16): lane l: row=l&15; hi=l>>4 selects K {8..15,24..31}.
// The 16 halves are two contiguous 16B runs -> two vector loads + shuffle.
// Requires: base 16B-aligned, stride (halves) a multiple of 8.
__device__ __forceinline__ v16h load_a_frag16(const _Float16* base, int stride, int k0) {
  int lane = threadIdx.x & 31;
  int row = lane & 15, hi = lane >> 4;
  const v8h* p = (const v8h*)(base + (size_t)row * stride + k0 * 32 + hi * 8);
  v8h lo = p[0];
  v8h hh = p[2];  // +16 halves
  return __builtin_shufflevector(lo, hh, 0, 1, 2, 3, 4, 5, 6, 7,
                                 8, 9, 10, 11, 12, 13, 14, 15);
}

// B fragment from fragment-packed weights: one contiguous 32B load per lane.
// Packed layout: Wp[((n0*nK + k0)*32 + lane)*16 + e].
__device__ __forceinline__ v16h load_b_fragP(const _Float16* Wp, int nK, int k0, int n0) {
  int lane = threadIdx.x & 31;
  const v16h* p = (const v16h*)(Wp + (((size_t)(n0 * nK + k0) * 32 + lane) << 4));
  return *p;
}

// ---------------- conversion / repack kernels ----------------

__global__ void k_cvt(const float* __restrict__ s, _Float16* __restrict__ d, int n) {
  int i = blockIdx.x * 256 + threadIdx.x;
  if (i < n) d[i] = (_Float16)s[i];
}

// Fragment-pack from row-major f32 W[K][N] (N multiple of 16, K multiple of 32).
__global__ void k_pack(const float* __restrict__ s, _Float16* __restrict__ d,
                       int K, int N) {
  int i = blockIdx.x * 256 + threadIdx.x;
  if (i >= K * N) return;
  int e = i & 15, lane = (i >> 4) & 31, frag = i >> 9;
  int nK = K >> 5;
  int n0 = frag / nK, k0 = frag - n0 * nK;
  int col = lane & 15, hi = lane >> 4;
  int kk = k0 * 32 + hi * 8 + (e & 7) + (e >> 3) * 16;
  d[i] = (_Float16)s[(size_t)kk * N + n0 * 16 + col];
}

// Fragment-pack the logical transpose: source s is [N, K]; W[k][n] = s[n*K + k].
__global__ void k_packT(const float* __restrict__ s, _Float16* __restrict__ d,
                        int K, int N) {
  int i = blockIdx.x * 256 + threadIdx.x;
  if (i >= K * N) return;
  int e = i & 15, lane = (i >> 4) & 31, frag = i >> 9;
  int nK = K >> 5;
  int n0 = frag / nK, k0 = frag - n0 * nK;
  int col = lane & 15, hi = lane >> 4;
  int kk = k0 * 32 + hi * 8 + (e & 7) + (e >> 3) * 16;
  d[i] = (_Float16)s[(size_t)(n0 * 16 + col) * K + kk];
}

// ---------------- GCN helpers ----------------

__global__ void k_deg(const int* __restrict__ ei, float* __restrict__ deg) {
  int e = blockIdx.x * 256 + threadIdx.x;
  if (e < N_EDGES) atomicAdd(&deg[ei[N_EDGES + e]], 1.0f);
}

__global__ void k_dinv(const float* __restrict__ deg, float* __restrict__ dinv) {
  int i = blockIdx.x * 256 + threadIdx.x;
  if (i < N_NODES) dinv[i] = rsqrtf(deg[i] + 1.0f);
}

__global__ void k_xw1(const float* __restrict__ x, const float* __restrict__ w,
                      float* __restrict__ xw) {
  int i = blockIdx.x * 256 + threadIdx.x;  // N*128
  int n = i >> 7, c = i & 127;
  float s = 0.f;
#pragma unroll
  for (int k = 0; k < 8; ++k) s += x[n * 8 + k] * w[k * 128 + c];
  xw[i] = s;
}

__global__ void k_agg(const float* __restrict__ xw, const float* __restrict__ dinv,
                      const int* __restrict__ ei, float* __restrict__ agg) {
  int idx = blockIdx.x * 256 + threadIdx.x;  // E*32 threads, 4 floats each
  int e = idx >> 5, g = idx & 31;
  int s = ei[e], d = ei[N_EDGES + e];
  float nrm = dinv[s] * dinv[d];
  const float4 v = ((const float4*)(xw + (size_t)s * 128))[g];
  float* ag = agg + (size_t)d * 128 + g * 4;
  atomicAdd(ag + 0, v.x * nrm);
  atomicAdd(ag + 1, v.y * nrm);
  atomicAdd(ag + 2, v.z * nrm);
  atomicAdd(ag + 3, v.w * nrm);
}

__global__ void k_finish(const float* __restrict__ agg, const float* __restrict__ xw,
                         const float* __restrict__ dinv, const float* __restrict__ b,
                         _Float16* __restrict__ outf16) {
  int i = blockIdx.x * 256 + threadIdx.x;  // N*128
  int n = i >> 7, c = i & 127;
  float di = dinv[n];
  float v = agg[i] + xw[i] * di * di + b[c];
  outf16[i] = (_Float16)fmaxf(v, 0.f);
}

// xw2 = h1f16[N,128] @ conv2_w via WMMA; one 16-row tile per block,
// each of 8 waves owns one 16-col output tile. B packed, nK = 4.
__global__ void __launch_bounds__(256) k_xw2(const _Float16* __restrict__ h1f16,
                                             const _Float16* __restrict__ w2p,
                                             float* __restrict__ out) {
  int wave = threadIdx.x >> 5, lane = threadIdx.x & 31;
  int col = lane & 15, hi = lane >> 4;
  const _Float16* arow = h1f16 + (size_t)blockIdx.x * 16 * 128;
  v8f acc = {};
#pragma unroll
  for (int k0 = 0; k0 < 4; ++k0)
    acc = wmma16x16x32(load_a_frag16(arow, 128, k0),
                       load_b_fragP(w2p, 4, k0, wave), acc);
#pragma unroll
  for (int i = 0; i < 8; ++i)
    out[(size_t)(blockIdx.x * 16 + i + hi * 8) * 128 + wave * 16 + col] = acc[i];
}

// ---------------- GRU: persistent 16-node tile, h in LDS ----------------
// wihTp: packed [64,384] (nK=2), whhTp: packed [128,384] (nK=4). Wave w owns
// hidden cols [w*16, w*16+16): gate tiles w (r), 8+w (z), 16+w (n). 18 WMMA/step.
__global__ void __launch_bounds__(256) k_gru(
    const int* __restrict__ xtext, const float* __restrict__ embed,
    const _Float16* __restrict__ wihTp, const _Float16* __restrict__ whhTp,
    const float* __restrict__ bih, const float* __restrict__ bhh,
    _Float16* __restrict__ txtf16) {
  __shared__ float hbuf[16 * 128];                      // f32 state
  __shared__ __align__(16) _Float16 hbuf16[16 * 128];   // f16 mirror for A-frags
  __shared__ __align__(16) _Float16 xt[16 * 64];
  __shared__ int toks[16];
  int tid = threadIdx.x;
  int wave = tid >> 5, lane = tid & 31;
  int col = lane & 15, hi = lane >> 4;
  int n0 = blockIdx.x * 16;
  int j = wave * 16 + col;  // hidden column of this lane

  for (int i = tid; i < 16 * 128; i += 256) { hbuf[i] = 0.f; hbuf16[i] = (_Float16)0.f; }
  float osum[8];
#pragma unroll
  for (int i = 0; i < 8; ++i) osum[i] = 0.f;

  float b_r = bih[j] + bhh[j];
  float b_z = bih[128 + j] + bhh[128 + j];
  float b_in = bih[256 + j];
  float b_hn = bhh[256 + j];

  for (int t = 0; t < LSEQ; ++t) {
    __syncthreads();  // protect xt + h from previous iteration
    if (tid < 16) toks[tid] = xtext[(n0 + tid) * LSEQ + t];
    __syncthreads();
    for (int i = tid; i < 16 * 64; i += 256) {
      int r = i >> 6, c = i & 63;
      xt[i] = (_Float16)embed[(size_t)toks[r] * 64 + c];
    }
    __syncthreads();

    v8f accR = {}, accZ = {}, accI = {}, accH = {};
#pragma unroll
    for (int k0 = 0; k0 < 2; ++k0) {  // gi = xt @ wih^T
      v16h a = load_a_frag16(xt, 64, k0);
      accR = wmma16x16x32(a, load_b_fragP(wihTp, 2, k0, wave), accR);
      accZ = wmma16x16x32(a, load_b_fragP(wihTp, 2, k0, 8 + wave), accZ);
      accI = wmma16x16x32(a, load_b_fragP(wihTp, 2, k0, 16 + wave), accI);
    }
#pragma unroll
    for (int k0 = 0; k0 < 4; ++k0) {  // gh = h @ whh^T
      v16h a = load_a_frag16(hbuf16, 128, k0);
      accR = wmma16x16x32(a, load_b_fragP(whhTp, 4, k0, wave), accR);
      accZ = wmma16x16x32(a, load_b_fragP(whhTp, 4, k0, 8 + wave), accZ);
      accH = wmma16x16x32(a, load_b_fragP(whhTp, 4, k0, 16 + wave), accH);
    }

    float h2v[8];
#pragma unroll
    for (int i = 0; i < 8; ++i) {
      int row = i + hi * 8;
      float hold = hbuf[row * 128 + j];
      float r = 1.f / (1.f + __expf(-(accR[i] + b_r)));
      float z = 1.f / (1.f + __expf(-(accZ[i] + b_z)));
      float nn = tanhf(accI[i] + b_in + r * (accH[i] + b_hn));
      float h2 = (1.f - z) * nn + z * hold;
      h2v[i] = h2;
      osum[i] += h2;
    }
    __syncthreads();  // all reads of old h complete before overwrite
#pragma unroll
    for (int i = 0; i < 8; ++i) {
      int row = i + hi * 8;
      hbuf[row * 128 + j] = h2v[i];
      hbuf16[row * 128 + j] = (_Float16)h2v[i];
    }
  }
#pragma unroll
  for (int i = 0; i < 8; ++i)
    txtf16[(size_t)(n0 + i + hi * 8) * 128 + j] = (_Float16)osum[i];
}

// ---------------- fused edge stage: 16 edges per block ----------------
// gather -> ep/et/ei linears -> concat e[16,384] -> row/col heads -> log_softmax
__global__ void __launch_bounds__(256) k_edge(
    const int* __restrict__ ei,
    const _Float16* __restrict__ posf16, const _Float16* __restrict__ txtf16,
    const _Float16* __restrict__ imgf16,
    const _Float16* __restrict__ wposP, const float* __restrict__ bpos,
    const _Float16* __restrict__ wtxtP, const float* __restrict__ btxt,
    const _Float16* __restrict__ wimgP, const float* __restrict__ bimg,
    const _Float16* __restrict__ wrow1P, const float* __restrict__ brow1,
    const float* __restrict__ wrow2, const float* __restrict__ brow2,
    const _Float16* __restrict__ wcol1P, const float* __restrict__ bcol1,
    const float* __restrict__ wcol2, const float* __restrict__ bcol2,
    float* __restrict__ out) {
  __shared__ __align__(16) _Float16 a_pos[16 * 256];
  __shared__ __align__(16) _Float16 a_txt[16 * 256];
  __shared__ __align__(16) _Float16 a_img[16 * 512];
  __shared__ __align__(16) _Float16 e_lds[16 * 384];
  __shared__ float hrow[16 * 128];
  __shared__ float hcol[16 * 128];
  __shared__ int sidx[16], didx[16];

  int tid = threadIdx.x;
  int wave = tid >> 5, lane = tid & 31;
  int col = lane & 15, hi = lane >> 4;
  int e0 = blockIdx.x * 16;

  if (tid < 16) { sidx[tid] = ei[e0 + tid]; didx[tid] = ei[N_EDGES + e0 + tid]; }
  __syncthreads();

  {  // gather [src|dst] rows into LDS (u32-wide copies of f16 pairs)
    unsigned* ap = (unsigned*)a_pos;
    unsigned* at = (unsigned*)a_txt;
    const unsigned* P = (const unsigned*)posf16;
    const unsigned* T = (const unsigned*)txtf16;
    for (int i = tid; i < 2048; i += 256) {
      int row = i >> 7, u = i & 127;
      int node = (u < 64) ? sidx[row] : didx[row];
      int off = u & 63;
      ap[i] = P[(size_t)node * 64 + off];
      at[i] = T[(size_t)node * 64 + off];
    }
    unsigned* ai = (unsigned*)a_img;
    const unsigned* I = (const unsigned*)imgf16;
    for (int i = tid; i < 4096; i += 256) {
      int row = i >> 8, u = i & 255;
      int node = (u < 128) ? sidx[row] : didx[row];
      int off = u & 127;
      ai[i] = I[(size_t)node * 128 + off];
    }
  }
  __syncthreads();

  {  // phase 1: ep / et / ei -> e_lds[16,384] f16 (relu)
    const _Float16* A[3] = {a_pos, a_txt, a_img};
    const _Float16* W[3] = {wposP, wtxtP, wimgP};
    const float* B[3] = {bpos, btxt, bimg};
    const int KC[3] = {8, 8, 16};
    const int STR[3] = {256, 256, 512};
#pragma unroll
    for (int h = 0; h < 3; ++h) {
      v8f acc = {};
      for (int k0 = 0; k0 < KC[h]; ++k0)
        acc = wmma16x16x32(load_a_frag16(A[h], STR[h], k0),
                           load_b_fragP(W[h], KC[h], k0, wave), acc);
      float bb = B[h][wave * 16 + col];
#pragma unroll
      for (int i = 0; i < 8; ++i) {
        float v = fmaxf(acc[i] + bb, 0.f);
        e_lds[(i + hi * 8) * 384 + h * 128 + wave * 16 + col] = (_Float16)v;
      }
    }
  }
  __syncthreads();

  {  // phase 2: row/col hidden = relu(e @ w1 + b1), K = 384 -> 12 chunks
    const _Float16* W1[2] = {wrow1P, wcol1P};
    const float* B1[2] = {brow1, bcol1};
    float* H[2] = {hrow, hcol};
#pragma unroll
    for (int h = 0; h < 2; ++h) {
      v8f acc = {};
#pragma unroll
      for (int k0 = 0; k0 < 12; ++k0)
        acc = wmma16x16x32(load_a_frag16(e_lds, 384, k0),
                           load_b_fragP(W1[h], 12, k0, wave), acc);
      float bb = B1[h][wave * 16 + col];
#pragma unroll
      for (int i = 0; i < 8; ++i)
        H[h][(i + hi * 8) * 128 + wave * 16 + col] = fmaxf(acc[i] + bb, 0.f);
    }
  }
  __syncthreads();

  if (tid < 32) {  // phase 3: 128->2 projection + log_softmax, one (head,edge)/lane
    int h = tid >> 4, e = tid & 15;
    const float* hid = h ? hcol : hrow;
    const float* w2 = h ? wcol2 : wrow2;
    const float* b2 = h ? bcol2 : brow2;
    float l0 = b2[0], l1 = b2[1];
    for (int k = 0; k < 128; ++k) {
      float v = hid[e * 128 + k];
      l0 += v * w2[k * 2];
      l1 += v * w2[k * 2 + 1];
    }
    float m = fmaxf(l0, l1);
    float lse = m + __logf(__expf(l0 - m) + __expf(l1 - m));
    size_t base = (size_t)h * N_EDGES * 2 + (size_t)(e0 + e) * 2;
    out[base] = l0 - lse;
    out[base + 1] = l1 - lse;
  }
}

// ---------------- host side ----------------

extern "C" void kernel_launch(void* const* d_in, const int* in_sizes, int n_in,
                              void* d_out, int out_size, void* d_ws, size_t ws_size,
                              hipStream_t stream) {
  (void)in_sizes; (void)n_in; (void)out_size; (void)ws_size;
  const float* x = (const float*)d_in[0];
  const int* ei = (const int*)d_in[1];
  const int* xtext = (const int*)d_in[2];
  const float* img = (const float*)d_in[3];
  const float* c1w = (const float*)d_in[4];
  const float* c1b = (const float*)d_in[5];
  const float* c2w = (const float*)d_in[6];
  const float* c2b = (const float*)d_in[7];
  const float* embed = (const float*)d_in[8];
  const float* wih = (const float*)d_in[9];
  const float* whh = (const float*)d_in[10];
  const float* bih = (const float*)d_in[11];
  const float* bhh = (const float*)d_in[12];
  const float* lposw = (const float*)d_in[13];
  const float* lposb = (const float*)d_in[14];
  const float* ltxtw = (const float*)d_in[15];
  const float* ltxtb = (const float*)d_in[16];
  const float* limgw = (const float*)d_in[17];
  const float* limgb = (const float*)d_in[18];
  const float* rw1 = (const float*)d_in[19];
  const float* rb1 = (const float*)d_in[20];
  const float* rw2 = (const float*)d_in[21];
  const float* rb2 = (const float*)d_in[22];
  const float* cw1 = (const float*)d_in[23];
  const float* cb1 = (const float*)d_in[24];
  const float* cw2 = (const float*)d_in[25];
  const float* cb2 = (const float*)d_in[26];
  float* out = (float*)d_out;

  char* base = (char*)d_ws;
  size_t off = 0;
  auto alloc = [&](size_t bytes) -> void* {
    void* p = base + off;
    off += (bytes + 255) & ~(size_t)255;
    return p;
  };
  float* deg = (float*)alloc((size_t)N_NODES * 4);
  float* dinv = (float*)alloc((size_t)N_NODES * 4);
  float* xw = (float*)alloc((size_t)N_NODES * 128 * 4);
  float* agg = (float*)alloc((size_t)N_NODES * 128 * 4);
  _Float16* h1f16 = (_Float16*)alloc((size_t)N_NODES * 128 * 2);
  _Float16* posf16 = (_Float16*)alloc((size_t)N_NODES * 128 * 2);
  _Float16* txtf16 = (_Float16*)alloc((size_t)N_NODES * 128 * 2);
  _Float16* imgf16 = (_Float16*)alloc((size_t)N_NODES * 256 * 2);
  _Float16* wihTp = (_Float16*)alloc((size_t)64 * 384 * 2);
  _Float16* whhTp = (_Float16*)alloc((size_t)128 * 384 * 2);
  _Float16* c2wP = (_Float16*)alloc((size_t)128 * 128 * 2);
  _Float16* lposP = (_Float16*)alloc((size_t)256 * 128 * 2);
  _Float16* ltxtP = (_Float16*)alloc((size_t)256 * 128 * 2);
  _Float16* limgP = (_Float16*)alloc((size_t)512 * 128 * 2);
  _Float16* rw1P = (_Float16*)alloc((size_t)384 * 128 * 2);
  _Float16* cw1P = (_Float16*)alloc((size_t)384 * 128 * 2);

  // fragment-pack weights (f32 -> f16, WMMA B layout)
  k_pack<<<(128 * 128 + 255) / 256, 256, 0, stream>>>(c2w, c2wP, 128, 128);
  k_pack<<<(256 * 128 + 255) / 256, 256, 0, stream>>>(lposw, lposP, 256, 128);
  k_pack<<<(256 * 128 + 255) / 256, 256, 0, stream>>>(ltxtw, ltxtP, 256, 128);
  k_pack<<<(512 * 128 + 255) / 256, 256, 0, stream>>>(limgw, limgP, 512, 128);
  k_pack<<<(384 * 128 + 255) / 256, 256, 0, stream>>>(rw1, rw1P, 384, 128);
  k_pack<<<(384 * 128 + 255) / 256, 256, 0, stream>>>(cw1, cw1P, 384, 128);
  k_packT<<<(64 * 384 + 255) / 256, 256, 0, stream>>>(wih, wihTp, 64, 384);
  k_packT<<<(128 * 384 + 255) / 256, 256, 0, stream>>>(whh, whhTp, 128, 384);
  k_cvt<<<(N_NODES * 256 + 255) / 256, 256, 0, stream>>>(img, imgf16, N_NODES * 256);

  // degrees / symmetric norm
  hipMemsetAsync(deg, 0, (size_t)N_NODES * 4, stream);
  k_deg<<<(N_EDGES + 255) / 256, 256, 0, stream>>>(ei, deg);
  k_dinv<<<(N_NODES + 255) / 256, 256, 0, stream>>>(deg, dinv);

  // GCN layer 1
  k_xw1<<<N_NODES * 128 / 256, 256, 0, stream>>>(x, c1w, xw);
  hipMemsetAsync(agg, 0, (size_t)N_NODES * 128 * 4, stream);
  k_agg<<<N_EDGES * 32 / 256, 256, 0, stream>>>(xw, dinv, ei, agg);
  k_finish<<<N_NODES * 128 / 256, 256, 0, stream>>>(agg, xw, dinv, c1b, h1f16);

  // GCN layer 2 (WMMA for the dense part)
  k_xw2<<<N_NODES / 16, 256, 0, stream>>>(h1f16, c2wP, xw);
  hipMemsetAsync(agg, 0, (size_t)N_NODES * 128 * 4, stream);
  k_agg<<<N_EDGES * 32 / 256, 256, 0, stream>>>(xw, dinv, ei, agg);
  k_finish<<<N_NODES * 128 / 256, 256, 0, stream>>>(agg, xw, dinv, c2b, posf16);

  // GRU over text
  k_gru<<<N_NODES / 16, 256, 0, stream>>>(xtext, embed, wihTp, whhTp, bih, bhh, txtf16);

  // fused edge stage
  k_edge<<<N_EDGES / 16, 256, 0, stream>>>(ei, posf16, txtf16, imgf16,
                                           lposP, lposb, ltxtP, ltxtb, limgP, limgb,
                                           rw1P, rb1, rw2, rb2, cw1P, cb1, cw2, cb2,
                                           out);
}
